// CCCNN_41137196761531
// MI455X (gfx1250) — matrix operand order repaired and checked by reading, not compiled, not run on hardware
//
#include <hip/hip_runtime.h>

typedef __attribute__((ext_vector_type(16))) _Float16 v16h;
typedef __attribute__((ext_vector_type(8)))  _Float16 v8h;
typedef __attribute__((ext_vector_type(8)))  float    v8f;

#define BCn    192     // B*C signal groups
#define K2c    32      // conv2 out channels
#define Lc     1024    // signal length
#define NLAG   2047    // 2V-1
#define PADh   240     // left zero pad (halves) for B-matrix negative shifts
#define SPLEN  1536    // padded signal buffer length in halves
#define CSPLIT 4       // channel-sum split factor (occupancy)
#define KPER   (K2c / CSPLIT)

__device__ __forceinline__ float silu_f(float v) {
    return v / (1.0f + __expf(-v));
}

// ---------------- Kernel A: fused conv1+SiLU -> conv2+SiLU, fp16 out ----------
__global__ __launch_bounds__(256) void conv_fused(
    const float* __restrict__ x,  const float* __restrict__ w1,
    const float* __restrict__ b1, const float* __restrict__ w2,
    const float* __restrict__ b2, _Float16* __restrict__ h2)
{
    __shared__ float xs[260];          // x[t0-2 .. t0+257]
    __shared__ float h1s[16 * 258];    // h1[c1][t0-1 .. t0+256]
    __shared__ float w1s[48], b1s[16], b2s[32], w2s[1536];
    const int tid = threadIdx.x;
    const int s   = blockIdx.x >> 2;        // signal 0..191
    const int t0  = (blockIdx.x & 3) << 8;  // t tile base

    if (tid < 48) w1s[tid] = w1[tid];
    if (tid < 16) b1s[tid] = b1[tid];
    if (tid < 32) b2s[tid] = b2[tid];
    for (int i = tid; i < 1536; i += 256) w2s[i] = w2[i];
    for (int j = tid; j < 260; j += 256) {
        int t = t0 - 2 + j;
        xs[j] = (t >= 0 && t < Lc) ? x[s * Lc + t] : 0.0f;
    }
    __syncthreads();

    for (int idx = tid; idx < 16 * 258; idx += 256) {
        int c1 = idx / 258, j = idx % 258;
        int t = t0 - 1 + j;
        float v = 0.0f;               // conv2's zero padding of h1 at t=-1,1024
        if (t >= 0 && t < Lc) {
            v = silu_f(w1s[c1*3+0]*xs[j] + w1s[c1*3+1]*xs[j+1] +
                       w1s[c1*3+2]*xs[j+2] + b1s[c1]);
        }
        h1s[c1 * 258 + j] = v;
    }
    __syncthreads();

    for (int idx = tid; idx < 32 * 256; idx += 256) {
        int c2 = idx >> 8, tt = idx & 255;
        float acc = b2s[c2];
        #pragma unroll
        for (int c1 = 0; c1 < 16; ++c1) {
            const float* w = &w2s[(c2 * 16 + c1) * 3];
            const float* h = &h1s[c1 * 258 + tt];
            acc += w[0]*h[0] + w[1]*h[1] + w[2]*h[2];
        }
        h2[(s * K2c + c2) * Lc + t0 + tt] = (_Float16)silu_f(acc);
    }
}

// -------- Kernel B: autocorrelation via v_wmma_f32_16x16x32_f16 --------------
// Block = (group g, channel split). Wave w owns lag block d0 = 256*w; tile
// element (M=i, N=n) is lag d0+i+16n:
//   ac[d0+i+16n] = sum_{t'} s[t'+d0+i] * s[t'-16n]
// A[i,k] = sp[PAD + t' + d0 + i], B[k,n] = sp[PAD + t' - 16n], sp zero padded.
// d0/nchunks are readfirstlane'd -> scalar K loop, EXEC all-1s at every wmma.
// Double-buffered LDS: the global_load for channel k+1 is issued before the
// WMMA loop of channel k, so its latency hides under matrix math; one barrier
// per channel (buffer parity guarantees no read/write overlap).
__global__ __launch_bounds__(128) void autocorr_wmma(
    const _Float16* __restrict__ h2, float* __restrict__ cc_part)
{
    __shared__ __align__(32) _Float16 sp[2 * SPLEN];
    const int tid   = threadIdx.x;
    const int g     = blockIdx.x >> 2;          // group 0..191
    const int split = blockIdx.x & 3;           // channel split 0..3
    const int k0    = split * KPER;
    const int wave  = tid >> 5;
    const int lane  = tid & 31;
    const int lo    = lane & 15;                // lane % 16
    const int hi    = lane >> 4;                // lane half
    // wave-uniform scalars -> SGPRs -> scalar loop, no EXEC masking
    const int d0      = __builtin_amdgcn_readfirstlane(wave << 8);
    const int nchunks = __builtin_amdgcn_readfirstlane(32 - (wave << 3));

    for (int i = tid; i < 2 * SPLEN; i += 128) sp[i] = (_Float16)0.0f;

    // A fragment (16-bit A 16x32): halves e=0..15 -> K = e + 16*hi;
    // M = lo.
    const int abase0 = PADh + d0 + lo + (hi << 4);
    // B fragment (16-bit B 32x16): halves e=0..15 -> K = e + 16*hi; N = lo.
    const int bbase0 = PADh + (hi << 4) - (lo << 4);   // 32B aligned

    const v8h* src = (const v8h*)(h2 + (size_t)(g * K2c + k0) * Lc);
    v8h r = src[tid];                    // preload channel k0 (16B/lane)
    __syncthreads();                     // zero-fill visible

    v8f c = {};
    for (int kk = 0; kk < KPER; ++kk) {
        const int bufo = (kk & 1) * SPLEN;
        _Float16* buf = sp + bufo;
        ((v8h*)(buf + PADh))[tid] = r;   // stage current channel
        if (kk + 1 < KPER)               // issue next channel's load now;
            r = src[(size_t)(kk + 1) * (Lc / 8) + tid];  // waits next iter
        __syncthreads();                 // staging visible to all waves

        const int ab0 = abase0 + bufo;
        const int bb0 = bbase0 + bufo;
        for (int kc = 0; kc < nchunks; ++kc) {
            v16h a, b;
            const int ab = ab0 + (kc << 5);
            #pragma unroll
            for (int e = 0; e < 8; ++e) a[e] = sp[ab + e];
            #pragma unroll
            for (int e = 0; e < 8; ++e) a[8 + e] = sp[ab + 8 + e];
            b = *(const v16h*)(sp + bb0 + (kc << 5));
            c = __builtin_amdgcn_wmma_f32_16x16x32_f16(
                    false, a, false, b, (short)0, c, false, false);
        }
        // no second barrier: next iteration writes the other buffer, and the
        // following barrier proves all waves left this buffer before reuse.
    }

    // C layout: VGPR r, lane -> M = r + 8*hi, N = lo  =>  d = d0 + M + 16*N
    float* row = cc_part + ((size_t)split * BCn + g) * NLAG;
    #pragma unroll
    for (int rr = 0; rr < 8; ++rr) {
        int d = d0 + rr + (hi << 3) + (lo << 4);
        float v = c[rr];
        row[1023 + d] = v;   // ac symmetric: ac[-d] = ac[d]
        row[1023 - d] = v;
    }
}

// -------- Kernel C1: sum channel-split partials + softmax + partial FC dot ----
__global__ __launch_bounds__(256) void softmax_fc_partial(
    const float* __restrict__ cc_part, const float* __restrict__ fc_w,
    float* __restrict__ rowdot)
{
    __shared__ float redA[256];
    __shared__ float redB[256];
    const int tid = threadIdx.x;
    const int g   = blockIdx.x;
    const int ch  = g % 3;
    const float* p0 = cc_part + ((size_t)0 * BCn + g) * NLAG;
    const float* p1 = cc_part + ((size_t)1 * BCn + g) * NLAG;
    const float* p2 = cc_part + ((size_t)2 * BCn + g) * NLAG;
    const float* p3 = cc_part + ((size_t)3 * BCn + g) * NLAG;
    const float* w0  = fc_w + 0 * (3 * NLAG) + ch * NLAG;
    const float* w1p = fc_w + 1 * (3 * NLAG) + ch * NLAG;

    float m = -3.0e38f;
    for (int j = tid; j < NLAG; j += 256)
        m = fmaxf(m, (p0[j] + p1[j]) + (p2[j] + p3[j]));
    redA[tid] = m; __syncthreads();
    for (int s = 128; s > 0; s >>= 1) {
        if (tid < s) redA[tid] = fmaxf(redA[tid], redA[tid + s]);
        __syncthreads();
    }
    m = redA[0]; __syncthreads();

    float sum = 0.0f, dot0 = 0.0f, dot1 = 0.0f;
    for (int j = tid; j < NLAG; j += 256) {
        float v = (p0[j] + p1[j]) + (p2[j] + p3[j]);
        float e = __expf(v - m);
        sum += e; dot0 += e * w0[j]; dot1 += e * w1p[j];
    }
    redA[tid] = sum; redB[tid] = dot0; __syncthreads();
    for (int s = 128; s > 0; s >>= 1) {
        if (tid < s) { redA[tid] += redA[tid + s]; redB[tid] += redB[tid + s]; }
        __syncthreads();
    }
    float tsum = redA[0], td0 = redB[0]; __syncthreads();
    redA[tid] = dot1; __syncthreads();
    for (int s = 128; s > 0; s >>= 1) {
        if (tid < s) redA[tid] += redA[tid + s];
        __syncthreads();
    }
    if (tid == 0) {
        rowdot[g * 2 + 0] = td0 / tsum;
        rowdot[g * 2 + 1] = redA[0] / tsum;
    }
}

// -------- Kernel C2: combine the 3 channel slices + bias ------------------
__global__ __launch_bounds__(128) void finalize_fc(
    const float* __restrict__ rowdot, const float* __restrict__ fc_b,
    float* __restrict__ out)
{
    const int tid = threadIdx.x;     // 128 = 64 batches * 2 outputs
    const int b = tid >> 1, o = tid & 1;
    float acc = fc_b[o];
    #pragma unroll
    for (int c = 0; c < 3; ++c) acc += rowdot[(b * 3 + c) * 2 + o];
    out[b * 2 + o] = acc;
}

extern "C" void kernel_launch(void* const* d_in, const int* in_sizes, int n_in,
                              void* d_out, int out_size, void* d_ws, size_t ws_size,
                              hipStream_t stream)
{
    const float* x    = (const float*)d_in[0];
    const float* w1   = (const float*)d_in[1];
    const float* b1   = (const float*)d_in[2];
    const float* w2   = (const float*)d_in[3];
    const float* b2   = (const float*)d_in[4];
    const float* fc_w = (const float*)d_in[5];
    const float* fc_b = (const float*)d_in[6];
    float* out = (float*)d_out;

    char* ws = (char*)d_ws;
    const size_t h2_bytes = (size_t)BCn * K2c * Lc * sizeof(_Float16);      // 12.6 MB
    const size_t cc_bytes = (size_t)CSPLIT * BCn * NLAG * sizeof(float);    // 6.29 MB
    _Float16* h2   = (_Float16*)ws;
    float*    ccp  = (float*)(ws + h2_bytes);
    float*    rdot = (float*)(ws + h2_bytes + cc_bytes);

    conv_fused        <<<BCn * 4,      256, 0, stream>>>(x, w1, b1, w2, b2, h2);
    autocorr_wmma     <<<BCn * CSPLIT, 128, 0, stream>>>(h2, ccp);
    softmax_fc_partial<<<BCn,          256, 0, stream>>>(ccp, fc_w, rdot);
    finalize_fc       <<<1,            128, 0, stream>>>(rdot, fc_b, out);
}
